// MDSFF_75453985457475
// MI455X (gfx1250) — compile-verified, matrix operand order checked by reading, not run on hardware
//
#include <hip/hip_runtime.h>
#include <hip/hip_bf16.h>
#include <math.h>

// Problem constants (fixed by the reference setup)
#define BATCH  4
#define CH     256
#define HH     64
#define WW     64
#define HWSZ   (HH * WW)        // 4096
#define KPTS   8
#define NHEADS 4
#define DHEAD  64

typedef __attribute__((ext_vector_type(16))) __bf16 v16bf;
typedef __attribute__((ext_vector_type(8)))  float  v8f;

// ---------------------------------------------------------------------------
// WMMA operand swizzles (CDNA5 16-bit layouts).
// Element e of a lane holds K = k0 + e + 8*((e>>3) + half).
// Inverse: for kk = k&31:  half = (kk>>3)&1,  e = (kk&7) | ((kk>>4)<<3).
// B slot layout: [kstep][lane = half*16 + n][e]   (one 32B run per lane/step)
// A slot layout: [mtile][kstep][lane = half*16 + (m&15)][e]
// ---------------------------------------------------------------------------
__device__ __forceinline__ int bslot(int k, int n) {
  int kk = k & 31;
  int half = (kk >> 3) & 1;
  int e = (kk & 7) | (((kk >> 4) & 1) << 3);
  return ((k >> 5) * 32 + half * 16 + n) * 16 + e;
}

// ---------------------------------------------------------------------------
// Kernel 0: pack the three 256x256 f32 weight matrices into bf16 in WMMA
// A-operand order (done once; 128KB each, stays L2-resident for all blocks).
// ---------------------------------------------------------------------------
__global__ void mdsff_pack_weights(const float* __restrict__ qw,
                                   const float* __restrict__ kw,
                                   const float* __restrict__ ow,
                                   __bf16* __restrict__ qa,
                                   __bf16* __restrict__ ka,
                                   __bf16* __restrict__ oa) {
  int idx = blockIdx.x * blockDim.x + threadIdx.x;  // m*256 + k
  int m = idx >> 8;
  int k = idx & (CH - 1);
  int kk = k & 31;
  int half = (kk >> 3) & 1;
  int e = (kk & 7) | (((kk >> 4) & 1) << 3);
  int slot = (((m >> 4) * 8 + (k >> 5)) * 32 + half * 16 + (m & 15)) * 16 + e;
  qa[slot] = (__bf16)qw[idx];
  ka[slot] = (__bf16)kw[idx];
  oa[slot] = (__bf16)ow[idx];
}

// Stage one 256(K) x 16(N) f32 tile (column base Bb, row stride HWSZ) into
// LDS as swizzled bf16. Coalesced float4 global reads; whole block helps.
__device__ __forceinline__ void stage_btile(const float* __restrict__ Bb,
                                            __bf16* __restrict__ btile) {
  for (int g = threadIdx.x * 4; g < CH * 16; g += blockDim.x * 4) {
    int k = g >> 4, n = g & 15;
    const float4 f = *(const float4*)(Bb + (size_t)k * HWSZ + n);
    btile[bslot(k, n + 0)] = (__bf16)f.x;
    btile[bslot(k, n + 1)] = (__bf16)f.y;
    btile[bslot(k, n + 2)] = (__bf16)f.z;
    btile[bslot(k, n + 3)] = (__bf16)f.w;
  }
}

// One wave: 16x16 f32 tile = packed-A-tile @ staged-B over K=256.
// Both operands are single contiguous 32B vector loads per step.
__device__ __forceinline__ v8f wmma_tile(const __bf16* __restrict__ Atile,
                                         const __bf16* __restrict__ btile,
                                         int lane) {
  v8f acc = {};
#pragma unroll
  for (int s = 0; s < 8; ++s) {
    v16bf av = *(const v16bf*)(Atile + (s * 32 + lane) * 16);
    v16bf bv = *(const v16bf*)(btile + (s * 32 + lane) * 16);
    acc = __builtin_amdgcn_wmma_f32_16x16x32_bf16(
        false, av, false, bv, (short)0, acc, false, false);
  }
  return acc;
}

// ---------------------------------------------------------------------------
// Kernel 1: 3x3 offset conv (256 -> 16 channels) + bias + tanh.
// ---------------------------------------------------------------------------
__global__ void mdsff_offset_conv(const float* __restrict__ x,
                                  const float* __restrict__ wgt,
                                  const float* __restrict__ bias,
                                  float* __restrict__ off) {
  int idx = blockIdx.x * blockDim.x + threadIdx.x;   // ((b*16+o)*H+h)*W+w
  int px = idx & (WW - 1);
  int py = (idx >> 6) & (HH - 1);
  int o  = (idx >> 12) & 15;
  int b  = idx >> 16;
  float acc = bias[o];
  const float* xb = x + (size_t)b * CH * HWSZ;
  const float* wo = wgt + (size_t)o * CH * 9;
  for (int c = 0; c < CH; ++c) {
    const float* xc = xb + (size_t)c * HWSZ;
    const float* wc = wo + c * 9;
#pragma unroll
    for (int dy = 0; dy < 3; ++dy) {
      int yy = py + dy - 1;
      if ((unsigned)yy >= HH) continue;
#pragma unroll
      for (int dx = 0; dx < 3; ++dx) {
        int xx = px + dx - 1;
        if ((unsigned)xx >= WW) continue;
        acc = fmaf(xc[yy * WW + xx], wc[dy * 3 + dx], acc);
      }
    }
  }
  off[idx] = tanhf(acc);
}

// ---------------------------------------------------------------------------
// Kernel 2: deformable bilinear sampling (zeros padding, align_corners=False).
// One thread per (b, k, hw); corner indices/weights computed once, reused for
// all 256 channels. global_prefetch_b8 a few channel-planes ahead to hide
// gather latency.
// ---------------------------------------------------------------------------
__global__ void mdsff_deform_sample(const float* __restrict__ xaux,
                                    const float* __restrict__ off,
                                    float* __restrict__ sampled) {
  int idx = blockIdx.x * blockDim.x + threadIdx.x;   // (b*KPTS+k)*HWSZ + hw
  int hw = idx & (HWSZ - 1);
  int kp = (idx >> 12) & (KPTS - 1);
  int b  = idx >> 15;
  int px = hw & (WW - 1);
  int py = hw >> 6;

  const float* offb = off + (size_t)b * (2 * KPTS) * HWSZ;
  float ox = offb[(2 * kp + 0) * HWSZ + hw];
  float oy = offb[(2 * kp + 1) * HWSZ + hw];

  float gx = (px + 0.5f) * (2.0f / WW) - 1.0f + ox * (2.0f / WW);
  float gy = (py + 0.5f) * (2.0f / HH) - 1.0f + oy * (2.0f / HH);
  gx = fminf(1.0f, fmaxf(-1.0f, gx));
  gy = fminf(1.0f, fmaxf(-1.0f, gy));

  float ix = ((gx + 1.0f) * WW - 1.0f) * 0.5f;
  float iy = ((gy + 1.0f) * HH - 1.0f) * 0.5f;
  float fx0 = floorf(ix), fy0 = floorf(iy);
  float wx = ix - fx0, wy = iy - fy0;
  int x0 = (int)fx0, y0 = (int)fy0;
  int x1 = x0 + 1, y1 = y0 + 1;
  float vx0 = ((unsigned)x0 < WW) ? 1.0f : 0.0f;
  float vx1 = ((unsigned)x1 < WW) ? 1.0f : 0.0f;
  float vy0 = ((unsigned)y0 < HH) ? 1.0f : 0.0f;
  float vy1 = ((unsigned)y1 < HH) ? 1.0f : 0.0f;
  float w00 = (1.0f - wx) * (1.0f - wy) * vx0 * vy0;
  float w10 = wx * (1.0f - wy) * vx1 * vy0;
  float w01 = (1.0f - wx) * wy * vx0 * vy1;
  float w11 = wx * wy * vx1 * vy1;
  int x0c = min(max(x0, 0), WW - 1), x1c = min(max(x1, 0), WW - 1);
  int y0c = min(max(y0, 0), HH - 1), y1c = min(max(y1, 0), HH - 1);
  int i00 = y0c * WW + x0c, i10 = y0c * WW + x1c;
  int i01 = y1c * WW + x0c, i11 = y1c * WW + x1c;

  const float* xb = xaux + (size_t)b * CH * HWSZ;
  float* outp = sampled + (size_t)(b * KPTS + kp) * CH * HWSZ + hw;
  for (int c = 0; c < CH; ++c) {
    const float* xc = xb + (size_t)c * HWSZ;
    __builtin_prefetch(xc + 4 * HWSZ + i00, 0, 1);
    __builtin_prefetch(xc + 4 * HWSZ + i11, 0, 1);
    float v = w00 * xc[i00] + w10 * xc[i10] + w01 * xc[i01] + w11 * xc[i11];
    outp[(size_t)c * HWSZ] = v;
  }
}

// ---------------------------------------------------------------------------
// Kernel 3/7: dense 256x256 GEMM per batch image:  D[b] = A @ B[b]
// Apack: bf16 weights pre-swizzled in WMMA A order. B,D: [BATCH][256][HWSZ].
// 512 threads = 16 waves; block stages one B tile in LDS (bf16, WMMA operand
// order), every wave (one M-tile) reuses it.
// ---------------------------------------------------------------------------
__global__ void __launch_bounds__(512) mdsff_gemm256(
    const __bf16* __restrict__ Apack, const float* __restrict__ B,
    float* __restrict__ D) {
  __shared__ __attribute__((aligned(32))) __bf16 btile[CH * 16];
  int lane = threadIdx.x & 31;
  int mt   = threadIdx.x >> 5;             // 0..15 -> M-tile
  int n0   = blockIdx.x * 16;
  int b    = blockIdx.z;
  int half = lane >> 4, mn = lane & 15;

  stage_btile(B + (size_t)b * CH * HWSZ + n0, btile);
  __syncthreads();

  v8f acc = wmma_tile(Apack + (size_t)mt * (8 * 32 * 16), btile, lane);

  float* Db = D + (size_t)b * CH * HWSZ;
#pragma unroll
  for (int r = 0; r < 8; ++r) {
    int m = mt * 16 + r + 8 * half;        // C/D layout: row = r + 8*half
    Db[(size_t)m * HWSZ + n0 + mn] = acc[r];
  }
}

// ---------------------------------------------------------------------------
// Kernel 4: fused k-projection + attention logits. Never materializes the
// 134MB k tensor: block (512 thr = 16 waves) handles (b, kp, 16-pixel tile);
// staged B tile shared by all 16 waves; wave wv computes channel tile
// c0=wv*16 of k_w@sampled via WMMA, dots against q, LDS-reduces per head.
// ---------------------------------------------------------------------------
__global__ void __launch_bounds__(512) mdsff_kproj_sim(
    const __bf16* __restrict__ Apack, const float* __restrict__ sampled,
    const float* __restrict__ q, float* __restrict__ sim) {
  __shared__ __attribute__((aligned(32))) __bf16 btile[CH * 16];
  __shared__ float simacc[NHEADS * 16];
  int lane = threadIdx.x & 31;
  int wv   = threadIdx.x >> 5;             // 0..15 -> M-tile (16 channels)
  int n0   = blockIdx.x * 16;
  int kp   = blockIdx.y;
  int b    = blockIdx.z;
  int half = lane >> 4, mn = lane & 15;

  if (threadIdx.x < NHEADS * 16) simacc[threadIdx.x] = 0.0f;
  stage_btile(sampled + (size_t)(b * KPTS + kp) * CH * HWSZ + n0, btile);
  __syncthreads();

  v8f acc = wmma_tile(Apack + (size_t)wv * (8 * 32 * 16), btile, lane);

  // q . k partial over this wave's 16 channels (all within one head)
  const float* qb = q + (size_t)b * CH * HWSZ + n0 + mn;
  float part = 0.0f;
#pragma unroll
  for (int r = 0; r < 8; ++r) {
    int c = wv * 16 + r + 8 * half;
    part = fmaf(acc[r], qb[(size_t)c * HWSZ], part);
  }
  int head = wv >> 2;                       // 4 waves (64 channels) per head
  atomicAdd(&simacc[head * 16 + mn], part);
  __syncthreads();

  if (threadIdx.x < NHEADS * 16) {
    int h = threadIdx.x >> 4, n = threadIdx.x & 15;
    sim[((size_t)(b * KPTS + kp) * NHEADS + h) * HWSZ + n0 + n] =
        simacc[threadIdx.x] * 0.125f;       // 1/sqrt(d), d=64
  }
}

// ---------------------------------------------------------------------------
// Kernel 5: softmax over K per head, then mean over heads -> w_k [B,K,HW].
// ---------------------------------------------------------------------------
__global__ void mdsff_softmax_mean(const float* __restrict__ sim,
                                   float* __restrict__ wk) {
  int idx = blockIdx.x * blockDim.x + threadIdx.x;  // b*HWSZ + hw
  int hw = idx & (HWSZ - 1);
  int b  = idx >> 12;
  float asum[KPTS];
#pragma unroll
  for (int k = 0; k < KPTS; ++k) asum[k] = 0.0f;
#pragma unroll
  for (int h = 0; h < NHEADS; ++h) {
    float v[KPTS];
    float m = -1e30f;
#pragma unroll
    for (int k = 0; k < KPTS; ++k) {
      v[k] = sim[((size_t)(b * KPTS + k) * NHEADS + h) * HWSZ + hw];
      m = fmaxf(m, v[k]);
    }
    float s = 0.0f;
#pragma unroll
    for (int k = 0; k < KPTS; ++k) { v[k] = __expf(v[k] - m); s += v[k]; }
    float inv = 1.0f / s;
#pragma unroll
    for (int k = 0; k < KPTS; ++k) asum[k] = fmaf(v[k], inv, asum[k]);
  }
#pragma unroll
  for (int k = 0; k < KPTS; ++k)
    wk[(size_t)(b * KPTS + k) * HWSZ + hw] = asum[k] * (1.0f / NHEADS);
}

// ---------------------------------------------------------------------------
// Kernel 6: out[b,c,hw] = sum_k sampled[b,k,c,hw] * w_k[b,k,hw].
// Second (last) full pass over `sampled`; should hit the 192MB L2.
// ---------------------------------------------------------------------------
__global__ void mdsff_weighted_sum(const float* __restrict__ sampled,
                                   const float* __restrict__ wk,
                                   float* __restrict__ out) {
  int idx = blockIdx.x * blockDim.x + threadIdx.x;  // (b*CH+c)*HWSZ + hw
  int hw = idx & (HWSZ - 1);
  int c  = (idx >> 12) & (CH - 1);
  int b  = idx >> 20;
  float acc = 0.0f;
#pragma unroll
  for (int k = 0; k < KPTS; ++k) {
    float s = sampled[((size_t)(b * KPTS + k) * CH + c) * HWSZ + hw];
    float w = wk[(size_t)(b * KPTS + k) * HWSZ + hw];
    acc = fmaf(s, w, acc);
  }
  out[idx] = acc;
}

// ---------------------------------------------------------------------------
// Launch. Inputs: 0:x_main 1:x_aux 2:offset_w 3:offset_b 4:q_w 5:k_w 6:out_w
// ---------------------------------------------------------------------------
extern "C" void kernel_launch(void* const* d_in, const int* in_sizes, int n_in,
                              void* d_out, int out_size, void* d_ws, size_t ws_size,
                              hipStream_t stream) {
  (void)in_sizes; (void)n_in; (void)out_size; (void)ws_size;
  const float* x_main   = (const float*)d_in[0];
  const float* x_aux    = (const float*)d_in[1];
  const float* offset_w = (const float*)d_in[2];
  const float* offset_b = (const float*)d_in[3];
  const float* q_w      = (const float*)d_in[4];
  const float* k_w      = (const float*)d_in[5];
  const float* out_w    = (const float*)d_in[6];

  float* ws = (float*)d_ws;
  // workspace layout (floats)
  float* sampled = ws;                                        // 33,554,432
  float* qbuf    = ws + (size_t)BATCH * KPTS * CH * HWSZ;     //  4,194,304
  float* simbuf  = qbuf + (size_t)BATCH * CH * HWSZ;          //    524,288
  float* wkbuf   = simbuf + (size_t)BATCH * KPTS * NHEADS * HWSZ; // 131,072
  float* outbuf  = wkbuf + (size_t)BATCH * KPTS * HWSZ;       //  4,194,304
  float* offbuf  = outbuf;   // conv output (262,144 fl) reuses out region
  // packed bf16 weights (WMMA A-operand order), 3 x 128KB
  __bf16* qa = (__bf16*)(outbuf + (size_t)BATCH * CH * HWSZ);
  __bf16* ka = qa + CH * CH;
  __bf16* oa = ka + CH * CH;

  // 0. pack weights to bf16 WMMA A layout (once per call, trivial cost)
  mdsff_pack_weights<<<dim3((CH * CH) / 256), dim3(256), 0, stream>>>(
      q_w, k_w, out_w, qa, ka, oa);
  // 1. offset conv + tanh
  mdsff_offset_conv<<<dim3((BATCH * 16 * HWSZ) / 256), dim3(256), 0, stream>>>(
      x_main, offset_w, offset_b, offbuf);
  // 2. deformable bilinear sampling
  mdsff_deform_sample<<<dim3((BATCH * KPTS * HWSZ) / 256), dim3(256), 0, stream>>>(
      x_aux, offbuf, sampled);
  // 3. q projection (WMMA, packed A + LDS-staged B)
  mdsff_gemm256<<<dim3(HWSZ / 16, 1, BATCH), dim3(512), 0, stream>>>(
      qa, x_main, qbuf);
  // 4. fused k projection + q.k logits (WMMA + LDS reduce)
  mdsff_kproj_sim<<<dim3(HWSZ / 16, KPTS, BATCH), dim3(512), 0, stream>>>(
      ka, sampled, qbuf, simbuf);
  // 5. softmax over K + head mean
  mdsff_softmax_mean<<<dim3((BATCH * HWSZ) / 256), dim3(256), 0, stream>>>(
      simbuf, wkbuf);
  // 6. weighted sum over K
  mdsff_weighted_sum<<<dim3((BATCH * CH * HWSZ) / 256), dim3(256), 0, stream>>>(
      sampled, wkbuf, outbuf);
  // 7. output projection (WMMA, packed A + LDS-staged B)
  mdsff_gemm256<<<dim3(HWSZ / 16, 1, BATCH), dim3(512), 0, stream>>>(
      oa, outbuf, (float*)d_out);
}
